// MoireLayer_77292231459671
// MI455X (gfx1250) — matrix operand order repaired and checked by reading, not compiled, not run on hardware
//
#include <hip/hip_runtime.h>
#include <hip/hip_bf16.h>
#include <stdint.h>

// MoireLayer for MI455X (gfx1250, wave32, WMMA f16 16x16x32).
// B=8, N=1024, DIN=DOUT=256, H=8, HD=32.

#define BB    8
#define NN    1024
#define DINC  256
#define DOUTC 256
#define NH    8
#define HDIM  32
#define MROWS (BB * NN)   // 8192

typedef _Float16 v16h __attribute__((ext_vector_type(16)));
typedef _Float16 h8   __attribute__((ext_vector_type(8)));
typedef float    v8f  __attribute__((ext_vector_type(8)));
typedef float    f4   __attribute__((ext_vector_type(4)));
typedef int      i4   __attribute__((ext_vector_type(4)));

// ---- optional gfx1250 async global->LDS path (guarded; falls back cleanly) ----
#if defined(__has_builtin)
#if __has_builtin(__builtin_amdgcn_global_load_async_to_lds_b128) && \
    __has_builtin(__builtin_amdgcn_s_wait_asynccnt)
#define USE_ASYNC_LDS 1
#endif
#endif
#ifndef USE_ASYNC_LDS
#define USE_ASYNC_LDS 0
#endif

static __device__ __forceinline__ v8f wmma_f16(v16h a, v16h b, v8f c) {
  // D = A(16x32 f16) * B(32x16 f16) + C(16x16 f32)
  return __builtin_amdgcn_wmma_f32_16x16x32_f16(
      /*neg_a=*/false, a, /*neg_b=*/false, b,
      /*c_mod=*/(short)0, c, /*reuse_a=*/false, /*reuse_b=*/false);
}

// A-fragment: 16x32 f16 tile, row-major source with leading dim lda.
// Lanes 0..15: M=lane, elements j -> K {0..7,16..23}; lanes 16..31: K {8..15,24..31}.
static __device__ __forceinline__ v16h load_a16(const _Float16* A, int lda,
                                                int row_base, int kbase, int lane) {
  const int m = lane & 15, kh = lane >> 4;
  const _Float16* p = A + (size_t)(row_base + m) * lda + kbase + kh * 8;
  h8 lo = *(const h8*)(p);        // K = kbase + kh*8 + 0..7
  h8 hi = *(const h8*)(p + 16);   // K = kbase + 16 + kh*8 + 0..7
  v16h r;
#pragma unroll
  for (int j = 0; j < 8; ++j) { r[j] = lo[j]; r[j + 8] = hi[j]; }
  return r;
}

// B-fragment: 32x16, B[k][n] = Bm[n][k] (Bm row-major f16, ldb).
// Lane L: N = L&15; lanes 0..15 hold K {0..7,16..23}, lanes 16..31 hold K {8..15,24..31}.
static __device__ __forceinline__ v16h load_b16(const _Float16* Bm, int ldb,
                                                int n_base, int kbase, int lane) {
  const int n = lane & 15, kh = lane >> 4;
  const _Float16* p = Bm + (size_t)(n_base + n) * ldb + kbase + kh * 8;
  h8 lo = *(const h8*)(p);
  h8 hi = *(const h8*)(p + 16);
  v16h r;
#pragma unroll
  for (int j = 0; j < 8; ++j) { r[j] = lo[j]; r[j + 8] = hi[j]; }
  return r;
}

// One wave computes a 16x64 strip of  A(f16, MxK) @ W^T (W pre-converted f16, NxK).
// A-fragment is loaded once per k-chunk and reused by 4 WMMAs.
static __device__ __forceinline__ void gemm4(const _Float16* A, const _Float16* Wh,
                                             int row_base, int n_base, int lane,
                                             v8f acc[4]) {
#pragma unroll
  for (int kc = 0; kc < 256; kc += 32) {
    v16h a = load_a16(A, 256, row_base, kc, lane);
#pragma unroll
    for (int t = 0; t < 4; ++t) {
      v16h b = load_b16(Wh, 256, n_base + t * 16, kc, lane);
      acc[t] = wmma_f16(a, b, acc[t]);
    }
  }
}

// ---------------- kernels ----------------

__global__ void cvt_f32_f16_kernel(const float* __restrict__ src,
                                   _Float16* __restrict__ dst, int n) {
  int i = blockIdx.x * blockDim.x + threadIdx.x;
  if (i < n) dst[i] = (_Float16)src[i];
}

// QKV projection: qkv = x @ Wqkv^T + bqkv ; scatter into Q (pre-scaled by 1/sqrt(HD)),
// K (B,H,N,HD) and V transposed (B,H,HD,N), all f16.
__global__ __launch_bounds__(32) void qkv_kernel(const _Float16* __restrict__ x16,
                                                 const _Float16* __restrict__ Wqkv16,
                                                 const float* __restrict__ bqkv,
                                                 _Float16* __restrict__ q,
                                                 _Float16* __restrict__ k,
                                                 _Float16* __restrict__ vt) {
  const int lane = threadIdx.x;
  const int row_base = blockIdx.x * 16;
  const int n_base   = blockIdx.y * 64;
  v8f acc[4] = {};
  gemm4(x16, Wqkv16, row_base, n_base, lane, acc);

#pragma unroll
  for (int t = 0; t < 4; ++t) {
    const int n = n_base + t * 16 + (lane & 15);   // 0..767
    const int which = n >> 8;                      // 0=Q 1=K 2=V
    const int dall = n & 255;
    const int head = dall >> 5, d = dall & 31;
    const float bv = bqkv[n];
#pragma unroll
    for (int v = 0; v < 8; ++v) {
      const int m = row_base + ((lane >> 4) * 8) + v;  // global row 0..8191
      const int b = m >> 10, pos = m & 1023;
      const int bh = b * NH + head;
      const float val = acc[t][v] + bv;
      if (which == 0)
        q[((size_t)bh * NN + pos) * HDIM + d] = (_Float16)(val * 0.17677669529663687f);
      else if (which == 1)
        k[((size_t)bh * NN + pos) * HDIM + d] = (_Float16)val;
      else
        vt[((size_t)bh * HDIM + d) * NN + pos] = (_Float16)val;
    }
  }
}

// Fused attention: one workgroup (256 thr / 8 waves) owns (b, h, 16-row block).
// Phase0: async-stage the contiguous 64KB adj strip adj[b, rb:rb+16, :] into LDS.
// Phase1: S = (Q K^T) * log_moire(adj) + 0.1*eye, masked; in-place over adj in LDS.
// Phase2: row softmax in LDS, P = exp(s - m) as f16 (16x1024).
// Phase3: O = P @ V with K-dim split across all 8 waves; LDS partial reduce;
//         normalize by row sum, write h (B,N,DOUT) f16.
__global__ __launch_bounds__(256) void attn_kernel(const _Float16* __restrict__ q,
                                                   const _Float16* __restrict__ kk,
                                                   const _Float16* __restrict__ vt,
                                                   const float* __restrict__ adj,
                                                   const uint8_t* __restrict__ mask,
                                                   const float* __restrict__ shifts,
                                                   const float* __restrict__ widths,
                                                   _Float16* __restrict__ hout) {
  extern __shared__ char smem[];
  float*    S    = (float*)smem;                               // 16*1024 f32 = 64KB (adj strip, then scores, then O-partials)
  _Float16* P    = (_Float16*)(smem + 16 * NN * 4);            // 16*1024 f16 = 32KB
  float*    red  = (float*)(smem + 16 * NN * 4 + 16 * NN * 2); // 256 f32
  float*    rowM = red + 256;                                  // 16
  float*    rowL = rowM + 16;                                  // 16

  const int rb = blockIdx.x * 16;
  const int h  = blockIdx.y;
  const int b  = blockIdx.z;
  const int bh = b * NH + h;
  const int tid = threadIdx.x;
  const int lane = tid & 31, wave = tid >> 5;

  const float shift = shifts[h];
  const float wdt   = widths[h];
  const float inv2w2 = 1.0f / (2.0f * wdt * wdt);
  const float LOGMIN = -20.723265836946414f;  // ln(1e-9)

  const _Float16* qb = q  + (size_t)bh * NN * HDIM;
  const _Float16* kb = kk + (size_t)bh * NN * HDIM;
  const _Float16* vb = vt + (size_t)bh * HDIM * NN;
  const uint8_t*  mrow = mask + (size_t)b * NN;

  // ---- Phase 0: stage adj[b, rb:rb+16, :] (contiguous 64KB) into LDS ----
  {
    const float* gsrc = adj + (size_t)b * NN * NN + (size_t)rb * NN;
#if USE_ASYNC_LDS
    // CDNA5 async global->LDS DMA path (ASYNCcnt); 16B per lane per issue.
    const char* gp = (const char*)gsrc;
    char* lp = (char*)S;
#pragma unroll 4
    for (int i = tid; i < (16 * NN * 4) / 16; i += 256) {
      __builtin_amdgcn_global_load_async_to_lds_b128(
          (__attribute__((address_space(1))) i4*)(gp + (size_t)i * 16),
          (__attribute__((address_space(3))) i4*)(lp + (size_t)i * 16),
          0, 0);
    }
    __builtin_amdgcn_s_wait_asynccnt(0);
#else
    f4* ldst = (f4*)S;
#pragma unroll 4
    for (int i = tid; i < (16 * NN) / 4; i += 256)
      ldst[i] = *(const f4*)(gsrc + (size_t)i * 4);
#endif
  }
  __syncthreads();

  // ---- Phase 1: scores, in place over the adj strip in LDS ----
  {
    v16h afrag = load_a16(qb, HDIM, rb, 0, lane);  // Q tile, K-dim = HD = 32 (one WMMA)
#pragma unroll 1
    for (int i = 0; i < 8; ++i) {
      const int ct = wave * 8 + i;                 // column tile 0..63
      v16h bfrag = load_b16(kb, HDIM, ct * 16, 0, lane);
      v8f c = {};
      c = wmma_f16(afrag, bfrag, c);
      const int col = ct * 16 + (lane & 15);
      const int mcol = mrow[col];
#pragma unroll
      for (int v = 0; v < 8; ++v) {
        const int lm = (lane >> 4) * 8 + v;
        const int row = rb + lm;
        const float a = S[lm * NN + col];          // adj from LDS
        const float diff = a - shift;
        float lmv = -(diff * diff) * inv2w2;       // log(exp(-t)) = -t
        lmv = fmaxf(lmv, LOGMIN);                  // clamp(moire, 1e-9)
        float s = c[v] * lmv + ((row == col) ? 0.1f : 0.0f);
        if (!(mrow[row] && mcol)) s = -1000.0f;
        S[lm * NN + col] = s;                      // same element, same thread: no race
      }
    }
  }
  __syncthreads();

  // ---- Phase 2: softmax over 1024 cols, 16 threads per row ----
  const int r = tid >> 4, sub = tid & 15;
  {
    float mx = -3.0e38f;
#pragma unroll 4
    for (int i = 0; i < 64; ++i) mx = fmaxf(mx, S[r * NN + sub * 64 + i]);
    red[r * 16 + sub] = mx;
  }
  __syncthreads();
  if (sub == 0) {
    float mx = red[r * 16];
    for (int j = 1; j < 16; ++j) mx = fmaxf(mx, red[r * 16 + j]);
    rowM[r] = mx;
  }
  __syncthreads();
  {
    const float mx = rowM[r];
    float l = 0.0f;
#pragma unroll 4
    for (int i = 0; i < 64; ++i) {
      const int cidx = sub * 64 + i;
      const float e = __expf(S[r * NN + cidx] - mx);
      P[r * NN + cidx] = (_Float16)e;
      l += e;
    }
    red[r * 16 + sub] = l;
  }
  __syncthreads();
  if (sub == 0) {
    float l = 0.0f;
    for (int j = 0; j < 16; ++j) l += red[r * 16 + j];
    rowL[r] = l;
  }
  __syncthreads();   // scores in S are dead past this point

  // ---- Phase 3: O = P @ V, K-dim (1024) split 128-wide across 8 waves ----
  float* Opart = S;  // 8 waves * 16 rows * 32 dims f32 = 16KB, reuses S
  {
    v8f acc0 = {}, acc1 = {};
    const int k0 = wave * 128;
#pragma unroll
    for (int i = 0; i < 4; ++i) {
      const int kc = k0 + i * 32;
      v16h a  = load_a16(P, NN, 0, kc, lane);              // from LDS
      v16h b0 = load_b16(vb, NN, 0,  kc, lane);            // V^T: contiguous K
      v16h b1 = load_b16(vb, NN, 16, kc, lane);
      acc0 = wmma_f16(a, b0, acc0);
      acc1 = wmma_f16(a, b1, acc1);
    }
    const int n_ = lane & 15;
#pragma unroll
    for (int v = 0; v < 8; ++v) {
      const int lm = (lane >> 4) * 8 + v;
      Opart[(wave * 16 + lm) * 32 + n_]      = acc0[v];
      Opart[(wave * 16 + lm) * 32 + 16 + n_] = acc1[v];
    }
  }
  __syncthreads();
  // reduce 8 partials: 512 outputs, 2 per thread
#pragma unroll
  for (int e = 0; e < 2; ++e) {
    const int idx = tid * 2 + e;       // 0..511
    const int lm = idx >> 5, d = idx & 31;
    float sum = 0.0f;
#pragma unroll
    for (int w2 = 0; w2 < 8; ++w2) sum += Opart[(w2 * 16 + lm) * 32 + d];
    const int row = rb + lm;
    float val = sum / rowL[lm];
    if (!mrow[row]) val = 0.0f;
    hout[((size_t)b * NN + row) * DOUTC + h * HDIM + d] = (_Float16)val;
  }
}

// FFN layer 1: h1 = relu(h @ W1^T + b1), f16 out.  16x64 per wave.
__global__ __launch_bounds__(32) void ffn1_kernel(const _Float16* __restrict__ hin,
                                                  const _Float16* __restrict__ W116,
                                                  const float* __restrict__ b1,
                                                  _Float16* __restrict__ h1) {
  const int lane = threadIdx.x;
  const int rb = blockIdx.x * 16, nb = blockIdx.y * 64;
  v8f acc[4] = {};
  gemm4(hin, W116, rb, nb, lane, acc);
#pragma unroll
  for (int t = 0; t < 4; ++t) {
    const int n = nb + t * 16 + (lane & 15);
    const float bv = b1[n];
#pragma unroll
    for (int v = 0; v < 8; ++v) {
      const int m = rb + ((lane >> 4) * 8) + v;
      h1[(size_t)m * DOUTC + n] = (_Float16)fmaxf(acc[t][v] + bv, 0.0f);
    }
  }
}

// x_proj = x @ Wp^T + bp, f32 out to workspace.  16x64 per wave.
__global__ __launch_bounds__(32) void xproj_kernel(const _Float16* __restrict__ x16,
                                                   const _Float16* __restrict__ Wp16,
                                                   const float* __restrict__ bp,
                                                   float* __restrict__ xp) {
  const int lane = threadIdx.x;
  const int rb = blockIdx.x * 16, nb = blockIdx.y * 64;
  v8f acc[4] = {};
  gemm4(x16, Wp16, rb, nb, lane, acc);
#pragma unroll
  for (int t = 0; t < 4; ++t) {
    const int n = nb + t * 16 + (lane & 15);
    const float bv = bp[n];
#pragma unroll
    for (int v = 0; v < 8; ++v) {
      const int m = rb + ((lane >> 4) * 8) + v;
      xp[(size_t)m * DOUTC + n] = acc[t][v] + bv;
    }
  }
}

// Final: out = rw * mask*(h1 @ W2^T + b2) + (1-rw) * x_proj, f32 out.  16x64 per wave.
__global__ __launch_bounds__(32) void final_kernel(const _Float16* __restrict__ h1,
                                                   const _Float16* __restrict__ W216,
                                                   const float* __restrict__ b2,
                                                   const float* __restrict__ xp,
                                                   const float* __restrict__ rwp,
                                                   const uint8_t* __restrict__ mask,
                                                   float* __restrict__ out) {
  const int lane = threadIdx.x;
  const int rb = blockIdx.x * 16, nb = blockIdx.y * 64;
  v8f acc[4] = {};
  gemm4(h1, W216, rb, nb, lane, acc);
  const float rw = rwp[0];
#pragma unroll
  for (int t = 0; t < 4; ++t) {
    const int n = nb + t * 16 + (lane & 15);
    const float bv = b2[n];
#pragma unroll
    for (int v = 0; v < 8; ++v) {
      const int m = rb + ((lane >> 4) * 8) + v;
      float hval = acc[t][v] + bv;
      if (!mask[m]) hval = 0.0f;
      out[(size_t)m * DOUTC + n] = rw * hval + (1.0f - rw) * xp[(size_t)m * DOUTC + n];
    }
  }
}

// ---------------- launch ----------------

extern "C" void kernel_launch(void* const* d_in, const int* in_sizes, int n_in,
                              void* d_out, int out_size, void* d_ws, size_t ws_size,
                              hipStream_t stream) {
  (void)in_sizes; (void)n_in; (void)out_size; (void)ws_size;
  const float*   x      = (const float*)d_in[0];
  const float*   adj    = (const float*)d_in[1];
  const uint8_t* mask   = (const uint8_t*)d_in[2];   // jnp bool -> 1 byte
  const float*   shifts = (const float*)d_in[3];
  const float*   widths = (const float*)d_in[4];
  const float*   Wqkv   = (const float*)d_in[5];
  const float*   bqkv   = (const float*)d_in[6];
  const float*   W1     = (const float*)d_in[7];
  const float*   b1     = (const float*)d_in[8];
  const float*   W2     = (const float*)d_in[9];
  const float*   b2     = (const float*)d_in[10];
  const float*   Wp     = (const float*)d_in[11];
  const float*   bp     = (const float*)d_in[12];
  const float*   rw     = (const float*)d_in[13];
  float* out = (float*)d_out;

  char* ws = (char*)d_ws;
  size_t off = 0;
  _Float16* x16  = (_Float16*)(ws + off); off += (size_t)MROWS * DINC * 2;        // 4MB
  _Float16* qf   = (_Float16*)(ws + off); off += (size_t)BB * NH * NN * HDIM * 2; // 4MB
  _Float16* kf   = (_Float16*)(ws + off); off += (size_t)BB * NH * NN * HDIM * 2; // 4MB
  _Float16* vtf  = (_Float16*)(ws + off); off += (size_t)BB * NH * HDIM * NN * 2; // 4MB
  _Float16* h16  = (_Float16*)(ws + off); off += (size_t)MROWS * DOUTC * 2;       // 4MB
  _Float16* h1f  = (_Float16*)(ws + off); off += (size_t)MROWS * DOUTC * 2;       // 4MB
  float*    xpf  = (float*)(ws + off);    off += (size_t)MROWS * DOUTC * 4;       // 8MB
  _Float16* wq16 = (_Float16*)(ws + off); off += (size_t)3 * DOUTC * DINC * 2;    // 384KB
  _Float16* w116 = (_Float16*)(ws + off); off += (size_t)DOUTC * DOUTC * 2;       // 128KB
  _Float16* w216 = (_Float16*)(ws + off); off += (size_t)DOUTC * DOUTC * 2;       // 128KB
  _Float16* wp16 = (_Float16*)(ws + off); off += (size_t)DOUTC * DINC * 2;        // 128KB

  // precision staging: x and all weights -> f16 once
  const int nx = MROWS * DINC;  // 2M
  cvt_f32_f16_kernel<<<nx / 256, 256, 0, stream>>>(x, x16, nx);
  cvt_f32_f16_kernel<<<(3 * DOUTC * DINC) / 256, 256, 0, stream>>>(Wqkv, wq16, 3 * DOUTC * DINC);
  cvt_f32_f16_kernel<<<(DOUTC * DOUTC) / 256, 256, 0, stream>>>(W1, w116, DOUTC * DOUTC);
  cvt_f32_f16_kernel<<<(DOUTC * DOUTC) / 256, 256, 0, stream>>>(W2, w216, DOUTC * DOUTC);
  cvt_f32_f16_kernel<<<(DOUTC * DINC) / 256, 256, 0, stream>>>(Wp, wp16, DOUTC * DINC);

  qkv_kernel<<<dim3(MROWS / 16, (3 * DOUTC) / 64), 32, 0, stream>>>(
      x16, wq16, bqkv, qf, kf, vtf);

  const size_t attn_smem = (size_t)16 * NN * 4 + (size_t)16 * NN * 2 + (256 + 32) * 4;
  attn_kernel<<<dim3(NN / 16, NH, BB), 256, attn_smem, stream>>>(
      qf, kf, vtf, adj, mask, shifts, widths, h16);

  ffn1_kernel<<<dim3(MROWS / 16, DOUTC / 64), 32, 0, stream>>>(h16, w116, b1, h1f);
  xproj_kernel<<<dim3(MROWS / 16, DOUTC / 64), 32, 0, stream>>>(x16, wp16, bp, xpf);
  final_kernel<<<dim3(MROWS / 16, DOUTC / 64), 32, 0, stream>>>(
      h1f, w216, b2, xpf, rw, mask, out);
}
